// GNNEncoder_40200893891315
// MI455X (gfx1250) — compile-verified, matrix operand order checked by reading, not run on hardware
//
#include <hip/hip_runtime.h>
#include <cstdint>
#include <cstddef>

// ---------------------------------------------------------------------------
// Types / constants
// ---------------------------------------------------------------------------
typedef __attribute__((ext_vector_type(16))) _Float16 v16h;
typedef __attribute__((ext_vector_type(8)))  float    v8f;

static constexpr int NCONS  = 20000;
static constexpr int NVARS  = 50000;
static constexpr int NEDGES = 500000;
static constexpr int D      = 128;
static constexpr float LN_EPS = 1e-5f;

// ---------------------------------------------------------------------------
// Workhorse WMMA GEMM:  C = act(A[N,K] @ W[K,M] + bias[M])
//   - one 128-row tile per block, 8 waves (wave32), wave w -> rows w*16..w*16+15
//   - each wave owns all M/16 column tiles; f32 accum (v8f per 16x16 tile)
//   - W staged once per block into LDS in *fragment-ready* per-lane order so
//     each lane reads its v16h B-fragment as contiguous 32B (2x ds_load_b128)
//   - A row index is CLAMPED (not branched): garbage only feeds rows the
//     guarded epilogue never stores -> branch-free k-loop.
//   - A loads are software-pipelined (double-buffered float4 staging) so the
//     next k-chunk's 4x global_load_b128 are in flight across the current
//     chunk's 8 WMMAs instead of serializing on one landing pad.
//   A fragment layout (16x32 f16, wave32): lane l (l&15 = row), khalf = l>>4,
//     elems 0..7  -> k = kc*32 + khalf*8 + j
//     elems 8..15 -> k = kc*32 + 16 + khalf*8 + (j-8)
//   B fragment symmetric with lane = column. C: vgpr r -> row r + (l>=16)*8,
//   col = tile*16 + (l&15).
// ---------------------------------------------------------------------------
template<int K, int M, int ACT>
__global__ __launch_bounds__(256) void k_gemm(const float* __restrict__ A,
                                              const float* __restrict__ W,
                                              const float* __restrict__ bias,
                                              float* __restrict__ C, int N) {
  constexpr int KC = K / 32;   // k-chunks of 32
  constexpr int MT = M / 16;   // 16-col tiles
  extern __shared__ _Float16 ldsW[];   // [KC][MT][32 lanes][16 halves]

  const int tid = threadIdx.x;

  // ---- stage W into fragment-ready LDS layout (f32 -> f16) ----
  for (int idx = tid; idx < K * M; idx += 256) {
    const int j  = idx & 15;
    const int l  = (idx >> 4) & 31;
    const int t  = (idx >> 9) % MT;
    const int kc = idx / (512 * MT);
    const int klocal = (j < 8 ? j : j + 8) + ((l >> 4) << 3);
    const int k = kc * 32 + klocal;
    const int n = t * 16 + (l & 15);
    ldsW[idx] = (_Float16)W[(size_t)k * M + n];
  }
  __syncthreads();

  const int wave    = tid >> 5;
  const int lane    = tid & 31;
  const int rowBase = blockIdx.x * 128 + wave * 16;
  const int khalf   = lane >> 4;
  // clamped A row: always a valid address, garbage only feeds guarded rows
  int arow = rowBase + (lane & 15);
  arow = arow < N ? arow : N - 1;
  const float* abase = A + (size_t)arow * K + khalf * 8;

  v8f acc[MT] = {};

  // double-buffered f32 staging for the A fragment (4x float4 per k-chunk)
  float4 bufA[4], bufB[4];
  {
    const float* ap = abase;
    bufA[0] = *(const float4*)(ap + 0);
    bufA[1] = *(const float4*)(ap + 4);
    bufA[2] = *(const float4*)(ap + 16);
    bufA[3] = *(const float4*)(ap + 20);
  }

#pragma unroll
  for (int kc = 0; kc < KC; ++kc) {
    float4* cur = (kc & 1) ? bufB : bufA;
    float4* nxt = (kc & 1) ? bufA : bufB;
    if (kc + 1 < KC) {
      const float* ap = abase + (kc + 1) * 32;
      nxt[0] = *(const float4*)(ap + 0);
      nxt[1] = *(const float4*)(ap + 4);
      nxt[2] = *(const float4*)(ap + 16);
      nxt[3] = *(const float4*)(ap + 20);
    }
    v16h a;
    a[0]  = (_Float16)cur[0].x; a[1]  = (_Float16)cur[0].y;
    a[2]  = (_Float16)cur[0].z; a[3]  = (_Float16)cur[0].w;
    a[4]  = (_Float16)cur[1].x; a[5]  = (_Float16)cur[1].y;
    a[6]  = (_Float16)cur[1].z; a[7]  = (_Float16)cur[1].w;
    a[8]  = (_Float16)cur[2].x; a[9]  = (_Float16)cur[2].y;
    a[10] = (_Float16)cur[2].z; a[11] = (_Float16)cur[2].w;
    a[12] = (_Float16)cur[3].x; a[13] = (_Float16)cur[3].y;
    a[14] = (_Float16)cur[3].z; a[15] = (_Float16)cur[3].w;
#pragma unroll
    for (int t = 0; t < MT; ++t) {
      const v16h b = *(const v16h*)(ldsW + ((size_t)(kc * MT + t) * 32 + lane) * 16);
      acc[t] = __builtin_amdgcn_wmma_f32_16x16x32_f16(
          /*neg_a=*/false, a, /*neg_b=*/false, b,
          /*c_mod=*/(short)0, acc[t], /*reuse_a=*/false, /*reuse_b=*/false);
    }
  }

  // ---- epilogue: bias + activation + store (row-guarded) ----
  const int rowOff = khalf * 8;
  const int nlo    = lane & 15;
#pragma unroll
  for (int t = 0; t < MT; ++t) {
    const int col = t * 16 + nlo;
    const float bv = bias ? bias[col] : 0.0f;
#pragma unroll
    for (int r = 0; r < 8; ++r) {
      const int row = rowBase + rowOff + r;
      if (row < N) {
        float v = acc[t][r] + bv;
        if (ACT) v = fmaxf(v, 0.0f);
        C[(size_t)row * M + col] = v;
      }
    }
  }
}

// ---------------------------------------------------------------------------
// LN(small feature dim) -> Linear(nf,128) -> ReLU.  One row per 128-thread
// block; nf <= 19 so each thread redundantly computes mean/var (L2-cached).
// ---------------------------------------------------------------------------
__global__ __launch_bounds__(128) void k_proj1(const float* __restrict__ X, int nf,
                                               const float* __restrict__ g,
                                               const float* __restrict__ b,
                                               const float* __restrict__ W1,
                                               const float* __restrict__ b1,
                                               float* __restrict__ out, int N) {
  const int row = blockIdx.x;
  if (row >= N) return;
  const int col = threadIdx.x;
  const float* x = X + (size_t)row * nf;
  float s = 0.f, ss = 0.f;
  for (int k = 0; k < nf; ++k) { const float v = x[k]; s += v; ss += v * v; }
  const float mean = s / nf;
  const float var  = ss / nf - mean * mean;
  const float inv  = rsqrtf(var + LN_EPS);
  float acc = b1[col];
  for (int k = 0; k < nf; ++k) {
    const float xn = (x[k] - mean) * inv * g[k] + b[k];
    acc += xn * W1[(size_t)k * D + col];
  }
  out[(size_t)row * D + col] = fmaxf(acc, 0.f);
}

// ---------------------------------------------------------------------------
// Per-edge: a = dstL[ci] + eW[e] + srcR[vi]; out = relu(LN(a)).
// One wave32 per edge row, 4 floats per lane; LN via __shfl_xor reduction.
// ---------------------------------------------------------------------------
__global__ __launch_bounds__(256) void k_edge_combine(
    const float* __restrict__ dstL, const float* __restrict__ eW,
    const float* __restrict__ srcR, const int* __restrict__ didx,
    const int* __restrict__ sidx, const float* __restrict__ g,
    const float* __restrict__ b, float* __restrict__ out, int nE) {
  const int e = blockIdx.x * 8 + (threadIdx.x >> 5);
  if (e >= nE) return;
  const int lane = threadIdx.x & 31;
  const int di = didx[e], si = sidx[e];
  const float4 x1 = *(const float4*)(dstL + (size_t)di * D + lane * 4);
  const float4 x2 = *(const float4*)(eW   + (size_t)e  * D + lane * 4);
  const float4 x3 = *(const float4*)(srcR + (size_t)si * D + lane * 4);
  float a0 = x1.x + x2.x + x3.x, a1 = x1.y + x2.y + x3.y;
  float a2 = x1.z + x2.z + x3.z, a3 = x1.w + x2.w + x3.w;
  float s  = a0 + a1 + a2 + a3;
  float ss = a0 * a0 + a1 * a1 + a2 * a2 + a3 * a3;
  for (int off = 16; off; off >>= 1) {
    s  += __shfl_xor(s, off, 32);
    ss += __shfl_xor(ss, off, 32);
  }
  const float mean = s * (1.f / D);
  const float var  = ss * (1.f / D) - mean * mean;
  const float inv  = rsqrtf(var + LN_EPS);
  const int c = lane * 4;
  float4 o;
  o.x = fmaxf((a0 - mean) * inv * g[c + 0] + b[c + 0], 0.f);
  o.y = fmaxf((a1 - mean) * inv * g[c + 1] + b[c + 1], 0.f);
  o.z = fmaxf((a2 - mean) * inv * g[c + 2] + b[c + 2], 0.f);
  o.w = fmaxf((a3 - mean) * inv * g[c + 3] + b[c + 3], 0.f);
  *(float4*)(out + (size_t)e * D + c) = o;
}

// ---------------------------------------------------------------------------
// zero-fill
// ---------------------------------------------------------------------------
__global__ __launch_bounds__(256) void k_zero(float* __restrict__ p, size_t n) {
  for (size_t i = (size_t)blockIdx.x * blockDim.x + threadIdx.x; i < n;
       i += (size_t)gridDim.x * blockDim.x)
    p[i] = 0.f;
}

// ---------------------------------------------------------------------------
// segment-sum via hardware fp32 atomics (global_atomic_add_f32)
// ---------------------------------------------------------------------------
__global__ __launch_bounds__(256) void k_scatter(const float* __restrict__ m,
                                                 const int* __restrict__ didx,
                                                 float* __restrict__ agg, int nE) {
  const long i = (long)blockIdx.x * blockDim.x + threadIdx.x;
  const int e = (int)(i >> 5);
  if (e >= nE) return;
  const int lane = (int)(i & 31);
  const float4 v = *(const float4*)(m + (size_t)e * D + lane * 4);
  float* p = agg + (size_t)didx[e] * D + lane * 4;
  unsafeAtomicAdd(p + 0, v.x);
  unsafeAtomicAdd(p + 1, v.y);
  unsafeAtomicAdd(p + 2, v.z);
  unsafeAtomicAdd(p + 3, v.w);
}

// ---------------------------------------------------------------------------
// hcat[row] = [ LN(agg[row]) , dst[row] ]   (row width 256)
// ---------------------------------------------------------------------------
__global__ __launch_bounds__(256) void k_concat_ln(
    const float* __restrict__ agg, const float* __restrict__ dst,
    const float* __restrict__ g, const float* __restrict__ b,
    float* __restrict__ out, int N) {
  const int row = blockIdx.x * 8 + (threadIdx.x >> 5);
  if (row >= N) return;
  const int lane = threadIdx.x & 31;
  const int c = lane * 4;
  const float4 a = *(const float4*)(agg + (size_t)row * D + c);
  float s  = a.x + a.y + a.z + a.w;
  float ss = a.x * a.x + a.y * a.y + a.z * a.z + a.w * a.w;
  for (int off = 16; off; off >>= 1) {
    s  += __shfl_xor(s, off, 32);
    ss += __shfl_xor(ss, off, 32);
  }
  const float mean = s * (1.f / D);
  const float var  = ss * (1.f / D) - mean * mean;
  const float inv  = rsqrtf(var + LN_EPS);
  float4 o;
  o.x = (a.x - mean) * inv * g[c + 0] + b[c + 0];
  o.y = (a.y - mean) * inv * g[c + 1] + b[c + 1];
  o.z = (a.z - mean) * inv * g[c + 2] + b[c + 2];
  o.w = (a.w - mean) * inv * g[c + 3] + b[c + 3];
  *(float4*)(out + (size_t)row * 256 + c) = o;
  const float4 d = *(const float4*)(dst + (size_t)row * D + c);
  *(float4*)(out + (size_t)row * 256 + D + c) = d;
}

// ---------------------------------------------------------------------------
// Host-side helpers
// ---------------------------------------------------------------------------
template<int K, int M, int ACT>
static inline void gemmL(const float* A, const float* W, const float* bias,
                         float* C, int N, hipStream_t s) {
  dim3 grid((unsigned)((N + 127) / 128));
  k_gemm<K, M, ACT><<<grid, dim3(256), K * M * 2, s>>>(A, W, bias, C, N);
}

struct Proj { const float *g, *b, *W1, *b1, *W2, *b2; };
struct Conv { const float *Wl, *bl, *We, *Wr, *lnf_g, *lnf_b, *Wf, *bf,
                          *lnp_g, *lnp_b, *Wo1, *bo1, *Wo2, *bo2; };
struct Lej  { const float *W1, *b1, *W2, *b2; };

extern "C" void kernel_launch(void* const* d_in, const int* in_sizes, int n_in,
                              void* d_out, int out_size, void* d_ws, size_t ws_size,
                              hipStream_t stream) {
  (void)in_sizes; (void)out_size; (void)ws_size;

  const float* consF = (const float*)d_in[0];   // [NCONS,5]
  const float* varF  = (const float*)d_in[1];   // [NVARS,19]
  const float* edgeF = (const float*)d_in[2];   // [NEDGES,8]
  const int*   eidx  = (const int*)d_in[n_in - 1]; // [2,NEDGES]
  const int*   ci = eidx;
  const int*   vi = eidx + NEDGES;

  // params flattened in insertion order starting at d_in[3]
  int pi = 3;
  auto getP = [&](void) { return (const float*)d_in[pi++]; };
  auto getProj = [&](void) {
    Proj p; p.g = getP(); p.b = getP(); p.W1 = getP(); p.b1 = getP();
    p.W2 = getP(); p.b2 = getP(); return p;
  };
  auto getConv = [&](void) {
    Conv c; c.Wl = getP(); c.bl = getP(); c.We = getP(); c.Wr = getP();
    c.lnf_g = getP(); c.lnf_b = getP(); c.Wf = getP(); c.bf = getP();
    c.lnp_g = getP(); c.lnp_b = getP(); c.Wo1 = getP(); c.bo1 = getP();
    c.Wo2 = getP(); c.bo2 = getP(); return c;
  };
  auto getLej = [&](void) {
    Lej l; l.W1 = getP(); l.b1 = getP(); l.W2 = getP(); l.b2 = getP(); return l;
  };
  const Proj pc = getProj(), pv = getProj(), pe = getProj();
  const Conv cv1 = getConv(), cv2 = getConv(), cv3 = getConv(), cv4 = getConv();
  const Lej  lc = getLej(), lv = getLej();

  // ---- workspace carve-out (floats) ----
  size_t off = 0;
  auto alloc = [&](size_t n) { float* p = (float*)d_ws + off; off += n; return p; };
  float* E    = alloc((size_t)NEDGES * D);   // projected edge features (persistent)
  float* BEW  = alloc((size_t)NEDGES * D);   // e@We per conv; also reused as m and proj temp
  float* BT   = alloc((size_t)NEDGES * D);   // relu(LN(a)) per conv; also node proj temp
  float* c0   = alloc((size_t)NCONS * D);
  float* c1   = alloc((size_t)NCONS * D);
  float* v0   = alloc((size_t)NVARS * D);
  float* v1   = alloc((size_t)NVARS * D);
  float* dstL = alloc((size_t)NVARS * D);
  float* srcR = alloc((size_t)NVARS * D);
  float* agg  = alloc((size_t)NVARS * D);
  float* hcat = alloc((size_t)NVARS * 256);
  float* h2   = alloc((size_t)NVARS * D);

  // ---- input projections ----
  k_proj1<<<dim3(NCONS), dim3(128), 0, stream>>>(consF, 5, pc.g, pc.b, pc.W1, pc.b1, BT, NCONS);
  gemmL<128, 128, 1>(BT, pc.W2, pc.b2, c0, NCONS, stream);
  k_proj1<<<dim3(NVARS), dim3(128), 0, stream>>>(varF, 19, pv.g, pv.b, pv.W1, pv.b1, BT, NVARS);
  gemmL<128, 128, 1>(BT, pv.W2, pv.b2, v0, NVARS, stream);
  k_proj1<<<dim3(NEDGES), dim3(128), 0, stream>>>(edgeF, 8, pe.g, pe.b, pe.W1, pe.b1, BEW, NEDGES);
  gemmL<128, 128, 1>(BEW, pe.W2, pe.b2, E, NEDGES, stream);

  // ---- one bipartite convolution ----
  auto run_conv = [&](const Conv& cp, const float* src, const float* dstIn,
                      float* dstOut, const int* sidx, const int* didx,
                      int nS, int nD) {
    gemmL<128, 128, 0>(dstIn, cp.Wl, cp.bl, dstL, nD, stream);          // left(x_i)
    gemmL<128, 128, 0>(src,   cp.Wr, nullptr, srcR, nS, stream);        // right(x_j)
    gemmL<128, 128, 0>(E,     cp.We, nullptr, BEW, NEDGES, stream);     // edge(e)
    k_edge_combine<<<dim3((NEDGES + 7) / 8), dim3(256), 0, stream>>>(
        dstL, BEW, srcR, didx, sidx, cp.lnf_g, cp.lnf_b, BT, NEDGES);   // relu(LN(a))
    gemmL<128, 128, 0>(BT, cp.Wf, cp.bf, BEW, NEDGES, stream);          // m
    k_zero<<<dim3(1024), dim3(256), 0, stream>>>(agg, (size_t)nD * D);
    k_scatter<<<dim3((NEDGES * 32 + 255) / 256), dim3(256), 0, stream>>>(
        BEW, didx, agg, NEDGES);                                        // segment_sum
    k_concat_ln<<<dim3((nD + 7) / 8), dim3(256), 0, stream>>>(
        agg, dstIn, cp.lnp_g, cp.lnp_b, hcat, nD);                      // [LN(agg), dst]
    gemmL<256, 128, 1>(hcat, cp.Wo1, cp.bo1, h2, nD, stream);
    gemmL<128, 128, 0>(h2, cp.Wo2, cp.bo2, dstOut, nD, stream);
  };

  // c = bconv(v->c); v = bconv(c->v); c = bconv(v->c); v = bconv(c->v)
  run_conv(cv1, v0, c0, c1, vi, ci, NVARS, NCONS);
  run_conv(cv2, c1, v0, v1, ci, vi, NCONS, NVARS);
  run_conv(cv3, v1, c1, c0, vi, ci, NVARS, NCONS);
  run_conv(cv4, c0, v1, v0, ci, vi, NCONS, NVARS);

  // ---- LEJ heads straight into d_out: [ce(20000,128); ve(50000,128)] ----
  float* out = (float*)d_out;
  gemmL<128, 64, 1>(c0, lc.W1, lc.b1, BT, NCONS, stream);
  gemmL<64, 128, 0>(BT, lc.W2, lc.b2, out, NCONS, stream);
  gemmL<128, 64, 1>(v0, lv.W1, lv.b1, BT, NVARS, stream);
  gemmL<64, 128, 0>(BT, lv.W2, lv.b2, out + (size_t)NCONS * D, NVARS, stream);
}
